// GraphCNN_79456894976125
// MI455X (gfx1250) — compile-verified, matrix operand order checked by reading, not run on hardware
//
#include <hip/hip_runtime.h>
#include <hip/hip_bf16.h>

// ---------------------------------------------------------------------------
// GIN forward for MI455X (gfx1250, wave32).
// GEMMs run on v_wmma_f32_16x16x32_bf16 (K=128 -> 4 WMMA steps); each wave
// computes a 32x64 tile (8 accumulators) so fragment-load latency is hidden
// behind two independent WMMA chains. Everything else is bandwidth-bound
// elementwise / atomic work.
// ---------------------------------------------------------------------------

typedef __attribute__((ext_vector_type(16))) __bf16 v16bf;
typedef __attribute__((ext_vector_type(8)))  float  v8f;

#define NN      50000
#define DD      128
#define EE      800000
#define GG      64
#define NLAYERS 4
#define FEA     (DD * (NLAYERS + 1))   // 640
#define BN_EPS  1e-5f

union Frag { uint4 u[2]; v16bf v; };

static __device__ __forceinline__ unsigned short f2bf(float x) {
    __hip_bfloat16 h = __float2bfloat16(x);
    union { __hip_bfloat16 h; unsigned short u; } cv; cv.h = h; return cv.u;
}

// ------------------------------- utility -----------------------------------

__global__ void zero_f32(float* p, int n) {
    int i = blockIdx.x * blockDim.x + threadIdx.x;
    if (i < n) p[i] = 0.0f;
}

__global__ void copy_f32(const float* __restrict__ src, float* __restrict__ dst, int n) {
    int i = blockIdx.x * blockDim.x + threadIdx.x;
    if (i < n) dst[i] = src[i];
}

__global__ void cvt_bf16(const float* __restrict__ src, unsigned short* __restrict__ dst, int n) {
    int i = blockIdx.x * blockDim.x + threadIdx.x;
    if (i < n) dst[i] = f2bf(src[i]);
}

// Pack the 8 [128x128] f32 weight matrices (W1[l], W2[l]) into per-lane WMMA
// B-fragment order: elem addr = mi*16384 + ((ntile*4 + kstep)*32 + lane)*16 + e
// lane: col = ntile*16 + (lane&15); kbase = kstep*32 + (lane>>4)*8;
// e<8 -> k = kbase+e ; e>=8 -> k = kbase+8+e (i.e. kbase+16+(e-8)).
__global__ void pack_weights(const float* __restrict__ W1, const float* __restrict__ W2,
                             unsigned short* __restrict__ Wpack) {
    int idx = blockIdx.x * blockDim.x + threadIdx.x;
    if (idx >= 8 * 16384) return;
    int mi   = idx >> 14;
    int e    = idx & 16383;
    int elem = e & 15;
    int lane = (e >> 4) & 31;
    int t    = (e >> 9) & 3;
    int s    = e >> 11;
    int col  = s * 16 + (lane & 15);
    int k    = t * 32 + ((lane >> 4) * 8) + (elem < 8 ? elem : elem + 8);
    int l    = mi >> 1;
    const float* W = (mi & 1) ? W2 : W1;
    Wpack[idx] = f2bf(W[((size_t)l * DD + k) * DD + col]);
}

// ------------------------------- graph ops ---------------------------------

// pooled = (1 + eps[l]) * h
__global__ void init_pooled(const float* __restrict__ h, const float* __restrict__ eps, int l,
                            float* __restrict__ pooled, int n) {
    int i = blockIdx.x * blockDim.x + threadIdx.x;
    if (i < n) pooled[i] = (1.0f + eps[l]) * h[i];
}

// pooled[row[e]] += h[col[e]]   (2 edges per 256-thread block)
__global__ void spmm_scatter(const float* __restrict__ h, const int* __restrict__ row,
                             const int* __restrict__ col, float* __restrict__ pooled, int E) {
    int e = blockIdx.x * 2 + (threadIdx.x >> 7);
    int d = threadIdx.x & 127;
    if (e < E)
        atomicAdd(&pooled[(size_t)row[e] * DD + d], h[(size_t)col[e] * DD + d]);
}

__global__ void count_nodes(const int* __restrict__ gid, float* __restrict__ counts, int N) {
    int i = blockIdx.x * blockDim.x + threadIdx.x;
    if (i < N) atomicAdd(&counts[gid[i]], 1.0f);
}

// gsum[gid[n], colOff + d] += H[n, d]
__global__ void pool_acc(const float* __restrict__ H, const int* __restrict__ gid,
                         float* __restrict__ gsum, int colOff, int N) {
    int n = blockIdx.x;
    int d = threadIdx.x;
    if (n < N)
        atomicAdd(&gsum[(size_t)gid[n] * FEA + colOff + d], H[(size_t)n * DD + d]);
}

// ------------------------------- GEMM (WMMA) -------------------------------

// C[N x 128] = A(bf16)[N x 128] @ Bpack + bias.  256 thr = 8 waves; each wave
// computes a 32(M) x 64(N) tile (8 accumulators); block covers 128 rows x 128
// cols.  Per K-step: 4 A loads + 8 B loads feed 8 independent WMMAs, with B
// fragments reused from registers across the two M sub-tiles.
__global__ void gemm_wmma_bf16(const unsigned short* __restrict__ A,
                               const unsigned short* __restrict__ Bpack,
                               const float* __restrict__ bias,
                               float* __restrict__ C, int N) {
    int wave = threadIdx.x >> 5;
    int lane = threadIdx.x & 31;
    int mwave = wave & 3;                      // 4 M-groups of 32 rows
    int nwave = wave >> 2;                     // 2 N-groups of 64 cols
    int rowBase = blockIdx.x * 128 + mwave * 32;
    int colBase = nwave * 64;

    int m    = lane & 15;
    int half = lane >> 4;
    int row0 = rowBase + m;
    int row1 = rowBase + 16 + m;
    int r0c  = row0 < N ? row0 : N - 1;        // clamp: keep EXEC all-1s
    int r1c  = row1 < N ? row1 : N - 1;

    v8f acc0[4] = {};
    v8f acc1[4] = {};

#pragma unroll
    for (int t = 0; t < 4; ++t) {
        int kb = t * 32 + half * 8;
        Frag fa0, fa1;
        fa0.u[0] = *reinterpret_cast<const uint4*>(A + (size_t)r0c * DD + kb);
        fa0.u[1] = *reinterpret_cast<const uint4*>(A + (size_t)r0c * DD + kb + 16);
        fa1.u[0] = *reinterpret_cast<const uint4*>(A + (size_t)r1c * DD + kb);
        fa1.u[1] = *reinterpret_cast<const uint4*>(A + (size_t)r1c * DD + kb + 16);
#pragma unroll
        for (int s = 0; s < 4; ++s) {
            int ntile = (colBase >> 4) + s;
            const uint4* pb = reinterpret_cast<const uint4*>(
                Bpack + ((((size_t)ntile * 4 + t) * 32 + lane) << 4));
            Frag fb;
            fb.u[0] = pb[0];
            fb.u[1] = pb[1];
            acc0[s] = __builtin_amdgcn_wmma_f32_16x16x32_bf16(
                false, fa0.v, false, fb.v, (short)0, acc0[s], false, false);
            acc1[s] = __builtin_amdgcn_wmma_f32_16x16x32_bf16(
                false, fa1.v, false, fb.v, (short)0, acc1[s], false, false);
        }
    }

#pragma unroll
    for (int s = 0; s < 4; ++s) {
        int c  = colBase + s * 16 + (lane & 15);
        float bv = bias[c];
#pragma unroll
        for (int j = 0; j < 8; ++j) {
            int r = rowBase + j + half * 8;
            if (r < N) C[(size_t)r * DD + c] = acc0[s][j] + bv;
            int r2 = r + 16;
            if (r2 < N) C[(size_t)r2 * DD + c] = acc1[s][j] + bv;
        }
    }
}

// ------------------------------- batch norm --------------------------------

#define STAT_ROWS 256

// stats[0..127] += column sums, stats[128..255] += column sums of squares
__global__ void colstats(const float* __restrict__ X, float* __restrict__ stats, int N) {
    int c  = threadIdx.x;                      // 128 threads
    int r0 = blockIdx.x * STAT_ROWS;
    int r1 = r0 + STAT_ROWS; if (r1 > N) r1 = N;
    float s = 0.0f, s2 = 0.0f;
    for (int r = r0; r < r1; ++r) {
        float v = X[(size_t)r * DD + c];
        s += v; s2 += v * v;
    }
    atomicAdd(&stats[c], s);
    atomicAdd(&stats[DD + c], s2);
}

// y = relu((x-m)*rsqrt(v+eps)*g + b); OUT_BF: write bf16 (next GEMM A), else f32
template <bool OUT_BF>
__global__ void bn_relu(const float* __restrict__ X, const float* __restrict__ stats,
                        const float* __restrict__ g, const float* __restrict__ b,
                        unsigned short* __restrict__ out_bf, float* __restrict__ out_f,
                        int n, float invN) {
    int i = blockIdx.x * blockDim.x + threadIdx.x;
    if (i >= n) return;
    int c = i & 127;
    float mean = stats[c] * invN;
    float var  = stats[DD + c] * invN - mean * mean;
    float y = (X[i] - mean) * rsqrtf(var + BN_EPS) * g[c] + b[c];
    y = y > 0.0f ? y : 0.0f;
    if (OUT_BF) out_bf[i] = f2bf(y);
    else        out_f[i]  = y;
}

// ------------------------------- head --------------------------------------

// rep[g, r] = sum_f (gsum[g,f]/count[g]) * Wpool[r,f]
__global__ void rep_kernel(const float* __restrict__ gsum, const float* __restrict__ counts,
                           const float* __restrict__ Wpool, float* __restrict__ rep) {
    int g = blockIdx.x;                        // 64
    int r = threadIdx.x;                       // 128
    float cnt = counts[g]; if (cnt < 1.0f) cnt = 1.0f;
    float inv = 1.0f / cnt;
    float acc = 0.0f;
    for (int f = 0; f < FEA; ++f)
        acc += gsum[(size_t)g * FEA + f] * inv * Wpool[(size_t)r * FEA + f];
    rep[(size_t)g * DD + r] = acc;
}

// score[g, o] = rep[g] . Wpred[o] + bpred[o]
__global__ void score_kernel(const float* __restrict__ rep, const float* __restrict__ Wpred,
                             const float* __restrict__ bpred, float* __restrict__ out) {
    int g = blockIdx.x;                        // 64
    int o = threadIdx.x;                       // 32, only 10 used
    if (o >= 10) return;
    float acc = bpred[o];
    for (int r = 0; r < DD; ++r)
        acc += rep[(size_t)g * DD + r] * Wpred[(size_t)o * DD + r];
    out[(size_t)g * 10 + o] = acc;
}

// ------------------------------- launch ------------------------------------

static inline size_t align256(size_t x) { return (x + 255) & ~(size_t)255; }

extern "C" void kernel_launch(void* const* d_in, const int* in_sizes, int n_in,
                              void* d_out, int out_size, void* d_ws, size_t ws_size,
                              hipStream_t stream) {
    const float* x      = (const float*)d_in[0];
    const int*   row    = (const int*)  d_in[1];
    const int*   col    = (const int*)  d_in[2];
    const int*   gid    = (const int*)  d_in[3];
    const float* eps    = (const float*)d_in[4];
    const float* W1     = (const float*)d_in[5];
    const float* b1     = (const float*)d_in[6];
    const float* g1     = (const float*)d_in[7];
    const float* be1    = (const float*)d_in[8];
    const float* W2     = (const float*)d_in[9];
    const float* b2     = (const float*)d_in[10];
    const float* bn_g   = (const float*)d_in[11];
    const float* bn_b   = (const float*)d_in[12];
    const float* Wpool  = (const float*)d_in[13];
    const float* Wpred  = (const float*)d_in[14];
    const float* bpred  = (const float*)d_in[15];
    float* out          = (float*)d_out;

    const int N = in_sizes[0] / DD;            // 50000
    const int E = in_sizes[1];                 // 800000
    const size_t ND = (size_t)N * DD;

    // workspace carve
    char* base = (char*)d_ws;
    size_t off = 0;
    float*          h      = (float*)(base + off);          off += align256(ND * 4);
    float*          pooled = (float*)(base + off);          off += align256(ND * 4);
    float*          t      = (float*)(base + off);          off += align256(ND * 4);
    unsigned short* abf    = (unsigned short*)(base + off); off += align256(ND * 2);
    unsigned short* Wpack  = (unsigned short*)(base + off); off += align256(8 * 16384 * 2);
    float*          stats  = (float*)(base + off);          off += align256(2 * DD * 4);
    float*          gsum   = (float*)(base + off);          off += align256((size_t)GG * FEA * 4);
    float*          counts = (float*)(base + off);          off += align256(GG * 4);
    float*          rep    = (float*)(base + off);          off += align256((size_t)GG * DD * 4);
    (void)ws_size;

    const int EW = 256;                                    // elementwise block
    const int nelemBlocks = (int)((ND + EW - 1) / EW);
    const int gemmBlocks  = (N + 127) / 128;
    const int statBlocks  = (N + STAT_ROWS - 1) / STAT_ROWS;
    const float invN = 1.0f / (float)N;

    // setup
    pack_weights<<<(8 * 16384 + EW - 1) / EW, EW, 0, stream>>>(W1, W2, Wpack);
    copy_f32<<<nelemBlocks, EW, 0, stream>>>(x, h, (int)ND);
    zero_f32<<<(GG * FEA + GG + EW - 1) / EW, EW, 0, stream>>>(gsum, GG * FEA + GG); // gsum+counts contiguous
    count_nodes<<<(N + EW - 1) / EW, EW, 0, stream>>>(gid, counts, N);
    pool_acc<<<N, DD, 0, stream>>>(x, gid, gsum, 0, N);

    for (int l = 0; l < NLAYERS; ++l) {
        const unsigned short* Bp1 = Wpack + (size_t)(l * 2 + 0) * 16384;
        const unsigned short* Bp2 = Wpack + (size_t)(l * 2 + 1) * 16384;

        init_pooled<<<nelemBlocks, EW, 0, stream>>>(h, eps, l, pooled, (int)ND);
        spmm_scatter<<<(E + 1) / 2, 256, 0, stream>>>(h, row, col, pooled, E);
        cvt_bf16<<<nelemBlocks, EW, 0, stream>>>(pooled, abf, (int)ND);

        gemm_wmma_bf16<<<gemmBlocks, 256, 0, stream>>>(abf, Bp1, b1 + l * DD, t, N);
        zero_f32<<<1, 256, 0, stream>>>(stats, 2 * DD);
        colstats<<<statBlocks, DD, 0, stream>>>(t, stats, N);
        bn_relu<true><<<nelemBlocks, EW, 0, stream>>>(t, stats, g1 + l * DD, be1 + l * DD,
                                                      abf, nullptr, (int)ND, invN);

        gemm_wmma_bf16<<<gemmBlocks, 256, 0, stream>>>(abf, Bp2, b2 + l * DD, t, N);
        zero_f32<<<1, 256, 0, stream>>>(stats, 2 * DD);
        colstats<<<statBlocks, DD, 0, stream>>>(t, stats, N);
        bn_relu<false><<<nelemBlocks, EW, 0, stream>>>(t, stats, bn_g + l * DD, bn_b + l * DD,
                                                       nullptr, h, (int)ND, invN);

        pool_acc<<<N, DD, 0, stream>>>(h, gid, gsum, DD * (l + 1), N);
    }

    rep_kernel<<<GG, DD, 0, stream>>>(gsum, counts, Wpool, rep);
    score_kernel<<<GG, 32, 0, stream>>>(rep, Wpred, bpred, out);
    (void)out_size; (void)n_in;
}
